// RGCN_30872224924290
// MI455X (gfx1250) — matrix operand order repaired and checked by reading, not compiled, not run on hardware
//
#include <hip/hip_runtime.h>
#include <hip/hip_bf16.h>

// ---------------------------------------------------------------------------
// Types / helpers
// ---------------------------------------------------------------------------
typedef __bf16 bf16;
typedef __attribute__((ext_vector_type(16))) __bf16 v16bf;
typedef __attribute__((ext_vector_type(8)))  float  v8f;
typedef unsigned int u32x4 __attribute__((ext_vector_type(4)));
typedef int          i32x4 __attribute__((ext_vector_type(4)));
typedef int          i32x8 __attribute__((ext_vector_type(8)));

__device__ __forceinline__ unsigned short f2bf_bits(float f) {
  unsigned int u = __builtin_bit_cast(unsigned int, f);
  u += 0x7fffu + ((u >> 16) & 1u);           // round-to-nearest-even
  return (unsigned short)(u >> 16);
}
__device__ __forceinline__ bf16 f2bf(float f) {
  return __builtin_bit_cast(bf16, f2bf_bits(f));
}
__device__ __forceinline__ float bf2f(bf16 h) {
  unsigned int u = ((unsigned int)__builtin_bit_cast(unsigned short, h)) << 16;
  return __builtin_bit_cast(float, u);
}

// ---------------------------------------------------------------------------
// TDM: DMA one 2D bf16 tile (rows x 64) from global into LDS.
// Tile row = 64 elems (128B); TDM pad inserts 16B per 128B -> LDS row stride
// 144B, matching the [72]-padded LDS arrays below. (ISA 08_async_tensor §8)
// ---------------------------------------------------------------------------
#define HAVE_TDM __has_builtin(__builtin_amdgcn_tensor_load_to_lds)

__device__ __forceinline__ void tdm_load_tile(unsigned lds_off, const bf16* gptr,
                                              int ld_elems, int rows) {
#if HAVE_TDM
  unsigned long long ga = (unsigned long long)gptr;
  u32x4 g0;
  g0[0] = 1u;                                              // count=1 (valid D#)
  g0[1] = lds_off;                                         // lds_addr
  g0[2] = (unsigned)ga;                                    // global_addr[31:0]
  g0[3] = (unsigned)((ga >> 32) & 0x1FFFFFFull) | (2u << 30); // addr[56:32] | type=2
  i32x8 g1;
  g1[0] = (int)((1u << 16)      // data_size = 2B
              | (1u << 20)      // pad_enable
              | (4u << 22)      // pad_interval: every 32 DWORDs (128B)
              | (3u << 25));    // pad_amount: 4 DWORDs (16B)
  g1[1] = (int)(((unsigned)ld_elems & 0xffffu) << 16);     // tensor_dim0 lo
  g1[2] = (int)((((unsigned)ld_elems) >> 16) | (((unsigned)rows & 0xffffu) << 16));
  g1[3] = (int)((((unsigned)rows) >> 16) | (64u << 16));   // tile_dim0 = 64
  g1[4] = rows;                                            // tile_dim1 (tile_dim2=0)
  g1[5] = ld_elems;                                        // tensor_dim0_stride lo
  g1[6] = 0;
  g1[7] = 0;
  i32x4 z4 = {0, 0, 0, 0};
#if defined(__clang_major__) && __clang_major__ >= 23
  i32x8 z8 = {0, 0, 0, 0, 0, 0, 0, 0};
  __builtin_amdgcn_tensor_load_to_lds(g0, g1, z4, z4, z8, 0);
#else
  __builtin_amdgcn_tensor_load_to_lds(g0, g1, z4, z4, 0);
#endif
#else
  (void)lds_off; (void)gptr; (void)ld_elems; (void)rows;
#endif
}

// ---------------------------------------------------------------------------
// fp32 -> bf16 convert (grid-stride)
// ---------------------------------------------------------------------------
__global__ __launch_bounds__(256) void cvt_f32_bf16(const float* __restrict__ in,
                                                    bf16* __restrict__ out, long n) {
  long i = (long)blockIdx.x * blockDim.x + threadIdx.x;
  long stride = (long)gridDim.x * blockDim.x;
  for (; i < n; i += stride) out[i] = f2bf(in[i]);
}

// ---------------------------------------------------------------------------
// Batched fp32 (R x Cc) -> bf16 same-layout (optional) + transposed (optional)
// block (32,8), grid (Cc/32, R/32, Batch)
// ---------------------------------------------------------------------------
__global__ __launch_bounds__(256) void trans_cvt_f32(const float* __restrict__ in,
                                                     bf16* __restrict__ outN,
                                                     bf16* __restrict__ outT,
                                                     int R, int Cc) {
  __shared__ bf16 tile[32][33];
  long b = blockIdx.z;
  const float* src = in + b * (long)R * Cc;
  int c0 = blockIdx.x * 32, r0 = blockIdx.y * 32;
  int tx = threadIdx.x, ty = threadIdx.y;
#pragma unroll
  for (int i = 0; i < 32; i += 8) {
    int r = r0 + ty + i, c = c0 + tx;
    bf16 h = f2bf(src[(long)r * Cc + c]);
    tile[ty + i][tx] = h;
    if (outN) outN[b * (long)R * Cc + (long)r * Cc + c] = h;
  }
  __syncthreads();
  if (outT) {
#pragma unroll
    for (int i = 0; i < 32; i += 8)
      outT[b * (long)R * Cc + (long)(c0 + ty + i) * R + (r0 + tx)] = tile[tx][ty + i];
  }
}

// ---------------------------------------------------------------------------
// Row softmax over N=256 fp32 logits -> bf16. grid (rows, batch), block 256
// ---------------------------------------------------------------------------
__global__ __launch_bounds__(256) void softmax_rows(const float* __restrict__ L,
                                                    bf16* __restrict__ S, int N) {
  __shared__ float red[256];
  long b = blockIdx.y;
  long row = blockIdx.x;
  const float* x = L + (b * N + row) * (long)N;
  int t = threadIdx.x;
  float v = x[t];
  red[t] = v; __syncthreads();
  for (int s = 128; s > 0; s >>= 1) { if (t < s) red[t] = fmaxf(red[t], red[t + s]); __syncthreads(); }
  float mx = red[0]; __syncthreads();
  float e = __expf(v - mx);
  red[t] = e; __syncthreads();
  for (int s = 128; s > 0; s >>= 1) { if (t < s) red[t] += red[t + s]; __syncthreads(); }
  float inv = 1.0f / red[0];
  S[(b * N + row) * (long)N + t] = f2bf(e * inv);
}

// ---------------------------------------------------------------------------
// Batched bf16 WMMA GEMM, fp32 accumulate. D(MxN) = A(MxK) * B
//   BT=1: B supplied transposed (N x K row-major)  -> TDM DMA staging
//   BT=0: B supplied normal     (K x N row-major)  -> per-thread transpose stage
// Block tile 128x128, BK=64, 256 threads (8 waves), wave tile 32x64,
// double-buffered LDS, TDM for A tiles (+B tiles when BT=1).
// MODE epilogues:
//   0: D->bf16   1: D->fp32   2: relu(D)->bf16
//   4: relu(D)+aux(bf16)->bf16   5: relu(D)+aux(bf16)->fp32   6: outF += relu(D)
// ---------------------------------------------------------------------------
template <int MODE, int BT>
__global__ __launch_bounds__(256) void gemm_wmma(
    const bf16* __restrict__ A, long sA,
    const bf16* __restrict__ Bm, long sB,
    int Nn, int Kk,
    const bf16* __restrict__ aux, long sAux,
    bf16* __restrict__ Ob, long sOb,
    float* __restrict__ Of, long sOf) {
  __shared__ bf16 lA[2][128][72];   // [m][k], row stride 144B (= TDM pad layout)
  __shared__ bf16 lB[2][128][72];   // [n][k]

  const int b = blockIdx.z;
  const int mBase = blockIdx.y * 128;
  const int nBase = blockIdx.x * 128;
  A  += (long)b * sA;
  Bm += (long)b * sB;

  const int tid   = threadIdx.x;
  const int lane  = tid & 31;
  const int wave  = tid >> 5;
  const int wr    = wave >> 1;     // 0..3 : 32-row strip
  const int wc    = wave & 1;      // 0..1 : 64-col strip
  const int lhalf = lane >> 4;
  const int l16   = lane & 15;

  // NN B staging map: thread handles 4 k-rows x 8 n-cols
  const int kq = (tid >> 4) * 4;   // 0..60
  const int nq = (tid & 15) * 8;   // 0..120

  const unsigned ldsA[2] = { (unsigned)(unsigned long long)(void*)&lA[0][0][0],
                             (unsigned)(unsigned long long)(void*)&lA[1][0][0] };
  const unsigned ldsB[2] = { (unsigned)(unsigned long long)(void*)&lB[0][0][0],
                             (unsigned)(unsigned long long)(void*)&lB[1][0][0] };

  v8f acc[2][4];
#pragma unroll
  for (int r = 0; r < 2; r++)
#pragma unroll
    for (int c = 0; c < 4; c++) acc[r][c] = 0.0f;

  const int nStages = Kk >> 6;

  uint4 r0, r1, r2, r3;   // NN staging registers

  auto issue_tdm = [&](int stg, int buf) {
#if HAVE_TDM
    tdm_load_tile(ldsA[buf], A + (long)mBase * Kk + stg * 64, Kk, 128);
    if (BT) tdm_load_tile(ldsB[buf], Bm + (long)nBase * Kk + stg * 64, Kk, 128);
#else
    // fallback: per-thread contiguous staging (thread: 1 row-half of 32 elems)
    {
      const int arow = tid >> 1, acol = (tid & 1) * 32;
      const bf16* g = A + (long)(mBase + arow) * Kk + stg * 64 + acol;
      uint4 q0 = *(const uint4*)(g);
      uint4 q1 = *(const uint4*)(g + 8);
      uint4 q2 = *(const uint4*)(g + 16);
      uint4 q3 = *(const uint4*)(g + 24);
      *(uint4*)(&lA[buf][arow][acol])      = q0;
      *(uint4*)(&lA[buf][arow][acol + 8])  = q1;
      *(uint4*)(&lA[buf][arow][acol + 16]) = q2;
      *(uint4*)(&lA[buf][arow][acol + 24]) = q3;
      if (BT) {
        const bf16* h = Bm + (long)(nBase + arow) * Kk + stg * 64 + acol;
        q0 = *(const uint4*)(h);
        q1 = *(const uint4*)(h + 8);
        q2 = *(const uint4*)(h + 16);
        q3 = *(const uint4*)(h + 24);
        *(uint4*)(&lB[buf][arow][acol])      = q0;
        *(uint4*)(&lB[buf][arow][acol + 8])  = q1;
        *(uint4*)(&lB[buf][arow][acol + 16]) = q2;
        *(uint4*)(&lB[buf][arow][acol + 24]) = q3;
      }
    }
#endif
  };

  auto loadB_nn = [&](int stg) {
    const bf16* g = Bm + (long)(stg * 64 + kq) * Nn + nBase + nq;
    r0 = *(const uint4*)(g);
    r1 = *(const uint4*)(g + Nn);
    r2 = *(const uint4*)(g + 2 * Nn);
    r3 = *(const uint4*)(g + 3 * Nn);
  };
  auto storeB_nn = [&](int buf) {
    union UU { uint4 u; unsigned short h[8]; };
    UU u0{r0}, u1{r1}, u2{r2}, u3{r3};
#pragma unroll
    for (int i = 0; i < 8; i++) {   // pack 4 k-values per column -> ds_store_b64
      uint2 w;
      w.x = (unsigned)u0.h[i] | ((unsigned)u1.h[i] << 16);
      w.y = (unsigned)u2.h[i] | ((unsigned)u3.h[i] << 16);
      *(uint2*)(&lB[buf][nq + i][kq]) = w;
    }
  };

  auto compute = [&](int buf) {
#pragma unroll
    for (int ks = 0; ks < 64; ks += 32) {
      v16bf af[2], bg[4];
#pragma unroll
      for (int r = 0; r < 2; r++) {
        union { v16bf v; uint4 u[2]; } t;
        const bf16* p = &lA[buf][wr * 32 + r * 16 + l16][ks];
        t.u[0] = *(const uint4*)(p + lhalf * 8);        // K 0..7 / 8..15
        t.u[1] = *(const uint4*)(p + 16 + lhalf * 8);   // K 16..23 / 24..31
        af[r] = t.v;
      }
#pragma unroll
      for (int c = 0; c < 4; c++) {
        union { v16bf v; uint4 u[2]; } t;
        const bf16* p = &lB[buf][wc * 64 + c * 16 + l16][ks + lhalf * 8];
        t.u[0] = *(const uint4*)(p);
        t.u[1] = *(const uint4*)(p + 16);
        bg[c] = t.v;
      }
#pragma unroll
      for (int r = 0; r < 2; r++)
#pragma unroll
        for (int c = 0; c < 4; c++)
          acc[r][c] = __builtin_amdgcn_wmma_f32_16x16x32_bf16(
              false, af[r], false, bg[c], (short)0, acc[r][c], false, false);
    }
  };

  // ---- prologue: stage 0 ----
#if HAVE_TDM
  if (wave == 0) issue_tdm(0, 0);
#else
  issue_tdm(0, 0);
#endif
  if constexpr (!BT) { loadB_nn(0); storeB_nn(0); }
  __builtin_amdgcn_s_wait_tensorcnt(0);
  __syncthreads();

  // ---- main loop, double buffered ----
  for (int stg = 0; stg < nStages; stg++) {
    const int cur = stg & 1, nxt = cur ^ 1;
    const bool more = (stg + 1) < nStages;
    if (more) {
#if HAVE_TDM
      if (wave == 0) issue_tdm(stg + 1, nxt);
#else
      issue_tdm(stg + 1, nxt);
#endif
      if constexpr (!BT) loadB_nn(stg + 1);
    }
    compute(cur);
    if (more) {
      if constexpr (!BT) storeB_nn(nxt);
      __builtin_amdgcn_s_wait_tensorcnt(0);
    }
    __syncthreads();
  }

  // ---- epilogue (ISA 16x16 f32 D layout: N=lane&15, M=vgpr + 8*(lane>=16)) ----
#pragma unroll
  for (int r = 0; r < 2; r++)
#pragma unroll
    for (int c = 0; c < 4; c++) {
      int gc0 = nBase + wc * 64 + c * 16 + l16;
#pragma unroll
      for (int i = 0; i < 8; i++) {
        int gr = mBase + wr * 32 + r * 16 + i + lhalf * 8;
        float v = acc[r][c][i];
        long idx = (long)gr * Nn + gc0;
        if (MODE == 0) Ob[(long)b * sOb + idx] = f2bf(v);
        else if (MODE == 1) Of[(long)b * sOf + idx] = v;
        else if (MODE == 2) Ob[(long)b * sOb + idx] = f2bf(fmaxf(v, 0.f));
        else if (MODE == 4) Ob[(long)b * sOb + idx] =
              f2bf(fmaxf(v, 0.f) + bf2f(aux[(long)b * sAux + idx]));
        else if (MODE == 5) Of[(long)b * sOf + idx] =
              fmaxf(v, 0.f) + bf2f(aux[(long)b * sAux + idx]);
        else if (MODE == 6) Of[(long)b * sOf + idx] += fmaxf(v, 0.f);
      }
    }
}

// ---------------------------------------------------------------------------
// Host-side pipeline
// ---------------------------------------------------------------------------
extern "C" void kernel_launch(void* const* d_in, const int* in_sizes, int n_in,
                              void* d_out, int out_size, void* d_ws, size_t ws_size,
                              hipStream_t stream) {
  (void)in_sizes; (void)n_in; (void)out_size; (void)ws_size;
  const int B = 64, N = 256, C = 512;
  const long NC  = (long)N * C;
  const long NN_ = (long)N * N;
  const long BNC = (long)B * NC;
  const long BNN = (long)B * NN_;
  const long CC  = (long)C * C;

  const float* f    = (const float*)d_in[0];
  const float* adj  = (const float*)d_in[2];
  const float* Wse1 = (const float*)d_in[3];
  const float* Wse2 = (const float*)d_in[4];
  const float* Wgc[9] = { (const float*)d_in[5],  (const float*)d_in[6],
                          (const float*)d_in[7],  (const float*)d_in[8],
                          (const float*)d_in[9],  (const float*)d_in[10],
                          (const float*)d_in[11], (const float*)d_in[12],
                          (const float*)d_in[13] };
  float* out = (float*)d_out;

  // workspace layout
  char* ws = (char*)d_ws;
  bf16* fb    = (bf16*)ws;  ws += BNC * 2;
  bf16* adjb  = (bf16*)ws;  ws += BNN * 2;
  bf16* adjT  = (bf16*)ws;  ws += BNN * 2;
  bf16* wse1b = (bf16*)ws;  ws += CC * 2;       // Wse1 as-is: already (out,in) = NT form
  bf16* wse2b = (bf16*)ws;  ws += CC * 2;
  bf16* wregT = (bf16*)ws;  ws += 9 * CC * 2;   // 9 GCN weights, pre-transposed to (out,in)
  bf16* bufA  = (bf16*)ws;  ws += BNC * 2;      // X1 / Y
  bf16* bufB  = (bf16*)ws;  ws += BNC * 2;      // X2 / Y
  bf16* bufC  = (bf16*)ws;  ws += BNC * 2;      // H / G
  bf16* Sb    = (bf16*)ws;  ws += BNN * 2;
  bf16* Tb    = (bf16*)ws;                      // relu branch temp (bf16)
  float* Lf   = (float*)Tb;                     // logits alias (same byte extent)

  const dim3 blk2d(32, 8);
  // ---- convert / pre-transpose inputs ----
  cvt_f32_bf16<<<2048, 256, 0, stream>>>(f, fb, BNC);
  trans_cvt_f32<<<dim3(N / 32, N / 32, B), blk2d, 0, stream>>>(adj, adjb, adjT, N, N);
  cvt_f32_bf16<<<512, 256, 0, stream>>>(Wse1, wse1b, CC);
  cvt_f32_bf16<<<512, 256, 0, stream>>>(Wse2, wse2b, CC);
  for (int i = 0; i < 9; i++)
    trans_cvt_f32<<<dim3(C / 32, C / 32, 1), blk2d, 0, stream>>>(
        Wgc[i], (bf16*)nullptr, wregT + i * CC, C, C);

  bf16* wst1  = wregT + 0 * CC; bf16* wst2  = wregT + 1 * CC; bf16* wst3  = wregT + 2 * CC;
  bf16* wst1b = wregT + 3 * CC; bf16* wst2b = wregT + 4 * CC; bf16* wst3b = wregT + 5 * CC;
  bf16* wsim1 = wregT + 6 * CC; bf16* wsim2 = wregT + 7 * CC; bf16* wsim3 = wregT + 8 * CC;

  const dim3 gF(C / 128, N / 128, B);   // 256x512 outputs : (4,2,64)
  const dim3 gL(N / 128, N / 128, B);   // 256x256 outputs : (2,2,64)

  // ---- similarity graph: S = softmax((f Wse1^T)(f Wse2^T)^T) ----
  gemm_wmma<0,1><<<gF, 256, 0, stream>>>(fb, NC, wse1b, 0, C, C, nullptr, 0, bufA, NC, nullptr, 0);
  gemm_wmma<0,1><<<gF, 256, 0, stream>>>(fb, NC, wse2b, 0, C, C, nullptr, 0, bufB, NC, nullptr, 0);
  gemm_wmma<1,1><<<gL, 256, 0, stream>>>(bufA, NC, bufB, NC, N, C, nullptr, 0, nullptr, 0, Lf, NN_);
  softmax_rows<<<dim3(N, B), 256, 0, stream>>>(Lf, Sb, N);

  // ---- st_GCN: three dual-branch layers ----
  gemm_wmma<0,1><<<gF, 256, 0, stream>>>(fb, NC, wst1,  0, C, C, nullptr, 0, bufA, NC, nullptr, 0);
  gemm_wmma<0,1><<<gF, 256, 0, stream>>>(fb, NC, wst1b, 0, C, C, nullptr, 0, bufB, NC, nullptr, 0);
  gemm_wmma<2,0><<<gF, 256, 0, stream>>>(adjb, NN_, bufA, NC, C, N, nullptr, 0, Tb, NC, nullptr, 0);
  gemm_wmma<4,0><<<gF, 256, 0, stream>>>(adjT, NN_, bufB, NC, C, N, Tb, NC, bufC, NC, nullptr, 0);

  gemm_wmma<0,1><<<gF, 256, 0, stream>>>(bufC, NC, wst2,  0, C, C, nullptr, 0, bufA, NC, nullptr, 0);
  gemm_wmma<0,1><<<gF, 256, 0, stream>>>(bufC, NC, wst2b, 0, C, C, nullptr, 0, bufB, NC, nullptr, 0);
  gemm_wmma<2,0><<<gF, 256, 0, stream>>>(adjb, NN_, bufA, NC, C, N, nullptr, 0, Tb, NC, nullptr, 0);
  gemm_wmma<4,0><<<gF, 256, 0, stream>>>(adjT, NN_, bufB, NC, C, N, Tb, NC, bufC, NC, nullptr, 0);

  gemm_wmma<0,1><<<gF, 256, 0, stream>>>(bufC, NC, wst3,  0, C, C, nullptr, 0, bufA, NC, nullptr, 0);
  gemm_wmma<0,1><<<gF, 256, 0, stream>>>(bufC, NC, wst3b, 0, C, C, nullptr, 0, bufB, NC, nullptr, 0);
  gemm_wmma<2,0><<<gF, 256, 0, stream>>>(adjb, NN_, bufA, NC, C, N, nullptr, 0, Tb, NC, nullptr, 0);
  gemm_wmma<5,0><<<gF, 256, 0, stream>>>(adjT, NN_, bufB, NC, C, N, Tb, NC, nullptr, 0, out, NC);

  // ---- sim_GCN: three layers; final accumulates into d_out ----
  gemm_wmma<0,1><<<gF, 256, 0, stream>>>(fb, NC, wsim1, 0, C, C, nullptr, 0, bufA, NC, nullptr, 0);
  gemm_wmma<2,0><<<gF, 256, 0, stream>>>(Sb, NN_, bufA, NC, C, N, nullptr, 0, bufC, NC, nullptr, 0);
  gemm_wmma<0,1><<<gF, 256, 0, stream>>>(bufC, NC, wsim2, 0, C, C, nullptr, 0, bufA, NC, nullptr, 0);
  gemm_wmma<2,0><<<gF, 256, 0, stream>>>(Sb, NN_, bufA, NC, C, N, nullptr, 0, bufC, NC, nullptr, 0);
  gemm_wmma<0,1><<<gF, 256, 0, stream>>>(bufC, NC, wsim3, 0, C, C, nullptr, 0, bufA, NC, nullptr, 0);
  gemm_wmma<6,0><<<gF, 256, 0, stream>>>(Sb, NN_, bufA, NC, C, N, nullptr, 0, nullptr, 0, out, NC);
}
